// CAM_Module_85641647882428
// MI455X (gfx1250) — matrix operand order repaired and checked by reading, not compiled, not run on hardware
//
#include <hip/hip_runtime.h>
#include <hip/hip_bf16.h>

// CAM module: B=16, C=128, H=W=128 (HW=16384), fp32 in/out.
// Pipeline: [memset energy] -> gram (bf16 WMMA, atomic f32 accumulate)
//           -> reversed softmax (writes bf16 attention)
//           -> attention x V (bf16 WMMA, async-LDS attention staging) + epilogue.

#define BATCH 16
#define CCH   128
#define HWSZ  16384

typedef __attribute__((ext_vector_type(16))) __bf16 v16bf;
typedef __attribute__((ext_vector_type(8)))  float  v8f;
typedef __attribute__((ext_vector_type(4)))  int    v4i;

#define GLOBAL_AS __attribute__((address_space(1)))
#define LDS_AS    __attribute__((address_space(3)))

union FragU { uint4 u[2]; v16bf v; };

#if defined(__has_builtin)
#if __has_builtin(__builtin_amdgcn_global_load_async_to_lds_b128)
#define HAVE_ASYNC 1
#endif
#endif
#ifndef HAVE_ASYNC
#define HAVE_ASYNC 0
#endif

#if HAVE_ASYNC
// 16-byte async copy global -> LDS (per-lane addresses). Signature per clang
// diagnostic: (v4i addrspace(1)*, v4i addrspace(3)*, imm offset, imm cpol).
__device__ __forceinline__ void async_cp16(const unsigned short* g, unsigned short* l) {
    GLOBAL_AS const unsigned short* g1 = (GLOBAL_AS const unsigned short*)g; // addrspacecast
    LDS_AS unsigned short*          l1 = (LDS_AS unsigned short*)l;          // addrspacecast
    __builtin_amdgcn_global_load_async_to_lds_b128(
        (GLOBAL_AS v4i*)g1, (LDS_AS v4i*)l1, 0, 0);
}
__device__ __forceinline__ void async_wait0() {
#if defined(__has_builtin) && __has_builtin(__builtin_amdgcn_s_wait_asynccnt)
    __builtin_amdgcn_s_wait_asynccnt(0);
#else
    asm volatile("s_wait_asynccnt 0" ::: "memory");
#endif
}
#endif

// f32 -> bf16 (round half up) pair pack: 2x v_add + 1x v_perm_b32.
__device__ __forceinline__ unsigned pack2(float a, float b) {
    unsigned ua = __builtin_bit_cast(unsigned, a) + 0x8000u;
    unsigned ub = __builtin_bit_cast(unsigned, b) + 0x8000u;
    return __builtin_amdgcn_perm(ub, ua, 0x07060302u);
}
__device__ __forceinline__ unsigned short f2bf(float f) {
    unsigned u = __builtin_bit_cast(unsigned, f) + 0x8000u;
    return (unsigned short)(u >> 16);
}

// ---------------------------------------------------------------------------
// Kernel 1: partial Gram  energy[b,i,j] += sum_k v[b,i,k]*v[b,j,k] over k-slab
// grid = (16 batches, 16 k-slabs), block = 256 (8 waves). Each wave owns a
// 2x4 block of 16x16 tiles (2 A frags + 4 B frags per k-step, 8 WMMAs, all
// fragments register-resident). LDS: double-buffered [128][40 bf16] chunks;
// chunk kt+1 global loads issue before the barrier (1 barrier/iteration).
// ---------------------------------------------------------------------------
__global__ __launch_bounds__(256) void gram_kernel(const float* __restrict__ x,
                                                   float* __restrict__ energy) {
    const int b    = blockIdx.x;
    const int ks   = blockIdx.y;
    const int tid  = threadIdx.x;
    const int lane = tid & 31;
    const int wave = tid >> 5;

    __shared__ unsigned short buf[2][128 * 40];   // bf16 [row][k], stride 40

    const float* xb = x + (size_t)b * CCH * HWSZ;
    const int k0slab = ks * (HWSZ / 16);          // 1024 K per slab

    v8f acc[8];                                   // [it*4 + jt]
#pragma unroll
    for (int t = 0; t < 8; ++t) acc[t] = (v8f){0.f,0.f,0.f,0.f,0.f,0.f,0.f,0.f};

    const int srow  = tid >> 1;      // staging: 2 threads per row, 16 floats each
    const int shalf = tid & 1;
    const int iblk  = (wave >> 1) * 2;   // wave's 2 M-tiles: iblk, iblk+1
    const int jblk  = (wave & 1) * 4;    // wave's 4 N-tiles: jblk..jblk+3
    const int hi    = lane >> 4;
    const int mr    = lane & 15;

    const float* sbase = xb + (size_t)srow * HWSZ + k0slab + shalf * 16;

    float4 st0, st1, st2, st3;
    {   // prologue: chunk 0 -> regs -> LDS buf[0]
        const float4* src = (const float4*)sbase;
        st0 = src[0]; st1 = src[1]; st2 = src[2]; st3 = src[3];
        unsigned* dst = (unsigned*)&buf[0][srow * 40 + shalf * 16];
        dst[0] = pack2(st0.x, st0.y); dst[1] = pack2(st0.z, st0.w);
        dst[2] = pack2(st1.x, st1.y); dst[3] = pack2(st1.z, st1.w);
        dst[4] = pack2(st2.x, st2.y); dst[5] = pack2(st2.z, st2.w);
        dst[6] = pack2(st3.x, st3.y); dst[7] = pack2(st3.z, st3.w);
    }

    for (int kt = 0; kt < 32; ++kt) {
        const int cur = kt & 1;
        if (kt + 1 < 32) {   // issue next chunk's global loads (latency overlap)
            const float4* src = (const float4*)(sbase + (kt + 1) * 32);
            st0 = src[0]; st1 = src[1]; st2 = src[2]; st3 = src[3];
        }
        __syncthreads();

        // ---- 2 A fragments (16x32 bf16 layout) ----
        FragU A[2];
#pragma unroll
        for (int it = 0; it < 2; ++it) {
            const unsigned short* pa = &buf[cur][((iblk + it) * 16 + mr) * 40 + hi * 8];
            A[it].u[0] = *(const uint4*)pa;          // K {0..7}/{8..15}
            A[it].u[1] = *(const uint4*)(pa + 16);   // K {16..23}/{24..31}
        }
        // ---- 4 B fragments (K-major, K {0..7}/{8..15} + {16..23}/{24..31}) ----
        FragU Bv[4];
#pragma unroll
        for (int jt = 0; jt < 4; ++jt) {
            const unsigned short* pb = &buf[cur][((jblk + jt) * 16 + mr) * 40 + hi * 8];
            Bv[jt].u[0] = *(const uint4*)pb;
            Bv[jt].u[1] = *(const uint4*)(pb + 16);
        }
        // ---- 8 WMMAs, all operands register-resident ----
#pragma unroll
        for (int it = 0; it < 2; ++it)
#pragma unroll
            for (int jt = 0; jt < 4; ++jt)
                acc[it * 4 + jt] = __builtin_amdgcn_wmma_f32_16x16x32_bf16(
                    false, A[it].v, false, Bv[jt].v, (short)0, acc[it * 4 + jt],
                    false, false);

        if (kt + 1 < 32) {   // convert + store next chunk into the other buffer
            unsigned* dst = (unsigned*)&buf[1 - cur][srow * 40 + shalf * 16];
            dst[0] = pack2(st0.x, st0.y); dst[1] = pack2(st0.z, st0.w);
            dst[2] = pack2(st1.x, st1.y); dst[3] = pack2(st1.z, st1.w);
            dst[4] = pack2(st2.x, st2.y); dst[5] = pack2(st2.z, st2.w);
            dst[6] = pack2(st3.x, st3.y); dst[7] = pack2(st3.z, st3.w);
        }
    }

    // ---- accumulate partial Gram into global energy (L2-resident 1 MB) ----
    const int moff = hi << 3;
#pragma unroll
    for (int it = 0; it < 2; ++it) {
#pragma unroll
        for (int jt = 0; jt < 4; ++jt) {
#pragma unroll
            for (int r = 0; r < 8; ++r) {
                size_t idx = (((size_t)b * CCH) + (iblk + it) * 16 + r + moff) * CCH
                           + (jblk + jt) * 16 + mr;
                unsafeAtomicAdd(&energy[idx], acc[it * 4 + jt][r]);
            }
        }
    }
}

// ---------------------------------------------------------------------------
// Kernel 2: reversed softmax; reads f32 energy, writes bf16 attention.
// softmax(max - e)_j == exp(min - e_j) / sum_j exp(min - e_j)
// One wave32 per row of 128; block = 8 waves.
// ---------------------------------------------------------------------------
__global__ __launch_bounds__(256) void softmax_kernel(const float* __restrict__ energy,
                                                      unsigned short* __restrict__ attbf) {
    const int row  = blockIdx.x * 8 + (threadIdx.x >> 5);   // 0..2047
    const int lane = threadIdx.x & 31;
    const float* p = energy + (size_t)row * CCH;
    unsigned short* o = attbf + (size_t)row * CCH;
    const float scale = 1.0f / 128.0f;                      // 1/sqrt(HW)

    float e[4];
#pragma unroll
    for (int q = 0; q < 4; ++q) e[q] = p[lane + 32 * q] * scale;

    float mn = fminf(fminf(e[0], e[1]), fminf(e[2], e[3]));
#pragma unroll
    for (int off = 16; off > 0; off >>= 1) mn = fminf(mn, __shfl_xor(mn, off, 32));

    float w[4], s = 0.f;
#pragma unroll
    for (int q = 0; q < 4; ++q) { w[q] = __expf(mn - e[q]); s += w[q]; }
#pragma unroll
    for (int off = 16; off > 0; off >>= 1) s += __shfl_xor(s, off, 32);

    const float inv = 1.0f / s;   // s >= 1 (exp(0) at the min element)
#pragma unroll
    for (int q = 0; q < 4; ++q) o[lane + 32 * q] = f2bf(w[q] * inv);
}

// ---------------------------------------------------------------------------
// Kernel 3: out[b,i,n] = gamma * (attention x V)[i,n] + x[b,i,n]
// grid = (256 n-chunks of 64, 16 batches), block = 256 (8 waves, wave = M tile).
// bf16 attention staged via GLOBAL_LOAD_ASYNC_TO_LDS_B128 into padded LDS;
// V chunk staged TRANSPOSED in LDS so B fragments are contiguous reads.
// ---------------------------------------------------------------------------
__global__ __launch_bounds__(256) void av_kernel(const float* __restrict__ x,
                                                 const unsigned short* __restrict__ attbf,
                                                 const float* __restrict__ gamma,
                                                 float* __restrict__ out) {
    const int nchunk = blockIdx.x;
    const int b      = blockIdx.y;
    const int tid    = threadIdx.x;
    const int lane   = tid & 31;
    const int wave   = tid >> 5;

    __shared__ unsigned short attbuf[128 * 136];   // bf16 [i][j], stride 136
    __shared__ unsigned short tbuf[64 * 136];      // bf16 [n][j]  (transposed V)

    const int nbase = nchunk * 64;
    const float* xb = x + (size_t)b * CCH * HWSZ;
    const unsigned short* abf = attbf + (size_t)b * CCH * CCH;

    // ---- stage bf16 attention: async global->LDS, 2 threads per row ----
    {
        const int row  = tid >> 1;
        const int half = tid & 1;
        const unsigned short* gsrc = abf + (size_t)row * CCH + half * 64;
        unsigned short* ldst = &attbuf[row * 136 + half * 64];
#pragma unroll
        for (int q = 0; q < 8; ++q) {
#if HAVE_ASYNC
            async_cp16(gsrc + q * 8, ldst + q * 8);
#else
            ((uint4*)ldst)[q] = ((const uint4*)gsrc)[q];
#endif
        }
    }
    // ---- stage V chunk transposed: coalesced read, scattered b16 LDS write ----
    {
        const int j  = tid >> 1;
        const int nh = (tid & 1) * 32;
        const float4* src = (const float4*)(xb + (size_t)j * HWSZ + nbase + nh);
#pragma unroll
        for (int q = 0; q < 8; ++q) {
            float4 f = src[q];
            const int n = nh + q * 4;
            tbuf[(n + 0) * 136 + j] = f2bf(f.x);
            tbuf[(n + 1) * 136 + j] = f2bf(f.y);
            tbuf[(n + 2) * 136 + j] = f2bf(f.z);
            tbuf[(n + 3) * 136 + j] = f2bf(f.w);
        }
    }
#if HAVE_ASYNC
    async_wait0();
#endif
    __syncthreads();

    const int ibase = wave * 16;
    const int hi    = lane >> 4;
    const int mr    = lane & 15;

    FragU A[4];                       // attention row fragments, K = 128 in 4 steps
#pragma unroll
    for (int kk = 0; kk < 4; ++kk) {
        const unsigned short* pa = &attbuf[(ibase + mr) * 136 + kk * 32 + hi * 8];
        A[kk].u[0] = *(const uint4*)pa;
        A[kk].u[1] = *(const uint4*)(pa + 16);
    }

    const float g = gamma[0];
#pragma unroll
    for (int nt = 0; nt < 4; ++nt) {
        v8f acc = (v8f){0.f,0.f,0.f,0.f,0.f,0.f,0.f,0.f};
        FragU Bv[2];   // 2-deep pipelined B fragments over the K chunks
        {
            const unsigned short* pb = &tbuf[(nt * 16 + mr) * 136 + hi * 8];
            Bv[0].u[0] = *(const uint4*)pb;
            Bv[0].u[1] = *(const uint4*)(pb + 16);
        }
#pragma unroll
        for (int kk = 0; kk < 4; ++kk) {
            if (kk + 1 < 4) {
                const unsigned short* pb =
                    &tbuf[(nt * 16 + mr) * 136 + (kk + 1) * 32 + hi * 8];
                Bv[(kk + 1) & 1].u[0] = *(const uint4*)pb;
                Bv[(kk + 1) & 1].u[1] = *(const uint4*)(pb + 16);
            }
            acc = __builtin_amdgcn_wmma_f32_16x16x32_bf16(
                false, A[kk].v, false, Bv[kk & 1].v, (short)0, acc, false, false);
        }
        const int col  = nbase + nt * 16 + mr;
        const int moff = hi << 3;
#pragma unroll
        for (int r = 0; r < 8; ++r) {
            size_t idx = ((size_t)b * CCH + ibase + r + moff) * (size_t)HWSZ + col;
            out[idx] = g * acc[r] + x[idx];
        }
    }
}

extern "C" void kernel_launch(void* const* d_in, const int* in_sizes, int n_in,
                              void* d_out, int out_size, void* d_ws, size_t ws_size,
                              hipStream_t stream) {
    const float* x     = (const float*)d_in[0];
    const float* gamma = (const float*)d_in[1];
    float* out    = (float*)d_out;
    float* energy = (float*)d_ws;                                     // 1 MB f32
    unsigned short* attbf =
        (unsigned short*)((char*)d_ws + (size_t)BATCH * CCH * CCH * 4); // 512 KB bf16

    (void)hipMemsetAsync(energy, 0, (size_t)BATCH * CCH * CCH * sizeof(float), stream);
    gram_kernel<<<dim3(BATCH, 16), 256, 0, stream>>>(x, energy);
    softmax_kernel<<<(BATCH * CCH) / 8, 256, 0, stream>>>(energy, attbf);
    av_kernel<<<dim3(HWSZ / 64, BATCH), 256, 0, stream>>>(x, attbf, gamma, out);
}
